// MDMDLSTMModel_59742995088042
// MI455X (gfx1250) — matrix operand order repaired and checked by reading, not compiled, not run on hardware
//
#include <hip/hip_runtime.h>

// ---------------------------------------------------------------------------
// MDMDLSTM on gfx1250 (MI455X):
//  * 127 anti-diagonal wavefront kernels, each cell = one 256-thread WG
//    computing a (32x288)@(288x640) bf16 WMMA GEMM + fused gate math.
//  * weights converted once to bf16 (stay resident in 192MB L2)
//  * h stored bf16 (feeds next cells' WMMA A-fragments + final conv)
//  * c kept in a 2-row rolling f32 buffer per direction (8.4 MB, L2-resident)
//  * final 1x1 conv = 131072x128x512 bf16 WMMA GEMM
// ---------------------------------------------------------------------------

typedef __attribute__((ext_vector_type(16))) __bf16 v16bf;
typedef __attribute__((ext_vector_type(8)))  __bf16 v8bf;
typedef __attribute__((ext_vector_type(8)))  float  v8f;

#define Bc   32
#define Cc   32
#define Hc   64
#define Wc   64
#define HDc  128
#define INc  288      // 2*HD + C
#define NG   640      // 5*HD
#define HW   4096     // Hc*Wc
#define OUTc 128
#define K4   512      // 4*HD

static __device__ __forceinline__ float sigm(float v) {
    return 1.0f / (1.0f + __expf(-v));
}

static __device__ __forceinline__ v16bf cat8(v8bf lo, v8bf hi) {
    return __builtin_shufflevector(lo, hi, 0,1,2,3,4,5,6,7,8,9,10,11,12,13,14,15);
}

// --- convert fp32 weights to bf16 once -------------------------------------
__global__ __launch_bounds__(256)
void prep_kernel(const float* __restrict__ Wl, const float* __restrict__ Wcv,
                 __bf16* __restrict__ Wlb, __bf16* __restrict__ Wcb) {
    int idx = blockIdx.x * 256 + threadIdx.x;
    if (idx < 4 * NG * INc) {
        Wlb[idx] = (__bf16)Wl[idx];
    } else {
        int k = idx - 4 * NG * INc;
        if (k < OUTc * K4) Wcb[k] = (__bf16)Wcv[k];
    }
}

// --- one anti-diagonal step: grid = (ncells, 4 dirs), block = 256 ----------
__global__ __launch_bounds__(256)
void diag_kernel(int d, int imin,
                 const float* __restrict__ x,
                 const __bf16* __restrict__ Wlb,
                 const float* __restrict__ bl,
                 const float* __restrict__ Wci, const float* __restrict__ Wf1,
                 const float* __restrict__ Wf2, const float* __restrict__ Wco,
                 __bf16* __restrict__ hbuf, float* __restrict__ cbuf) {
    __shared__ __align__(32) __bf16 lA[Bc * INc];   // 18 KB

    const int i   = imin + (int)blockIdx.x;
    const int j   = d - i;
    const int dir = blockIdx.y;
    const int f0  = dir & 1;            // flip over H
    const int f1  = (dir >> 1) & 1;     // flip over W
    const int ii  = f0 ? (Hc - 1 - i) : i;
    const int jj  = f1 ? (Wc - 1 - j) : j;
    const int tid = threadIdx.x;

    // ---- phase 1: stage A = [x_ij | h_up | h_left] (32 x 288) bf16 in LDS
    const __bf16* hdir = hbuf + (size_t)dir * HW * Bc * HDc;
    for (int idx = tid; idx < Bc * INc; idx += 256) {
        int b = idx / INc;
        int k = idx - b * INc;
        float v = 0.0f;
        if (k < Cc) {
            v = x[((b * Cc + k) * Hc + ii) * Wc + jj];
        } else if (k < Cc + HDc) {
            if (i > 0) v = (float)hdir[((size_t)((i - 1) * Wc + j) * Bc + b) * HDc + (k - Cc)];
        } else {
            if (j > 0) v = (float)hdir[((size_t)(i * Wc + (j - 1)) * Bc + b) * HDc + (k - Cc - HDc)];
        }
        lA[idx] = (__bf16)v;
    }
    __syncthreads();

    // ---- phase 2: GEMM. wave w owns channel slice [w*16, w*16+16) of each
    // of the 5 gates (nt = g*8 + w), so gate math stays wave-local.
    const int lane = tid & 31;
    const int wave = tid >> 5;
    const int lrow = lane & 15;
    const int lg   = lane >> 4;

    v8f acc[2][5];
#pragma unroll
    for (int m = 0; m < 2; ++m)
#pragma unroll
        for (int g = 0; g < 5; ++g) acc[m][g] = (v8f)0.0f;

    const __bf16* wbase = Wlb + (size_t)dir * NG * INc;
#pragma unroll
    for (int kt = 0; kt < 9; ++kt) {             // K = 288 = 9 * 32
        const int kb = kt * 32 + lg * 8;
        v8bf a0l = *(const v8bf*)&lA[(0  + lrow) * INc + kb];
        v8bf a0h = *(const v8bf*)&lA[(0  + lrow) * INc + kb + 16];
        v8bf a1l = *(const v8bf*)&lA[(16 + lrow) * INc + kb];
        v8bf a1h = *(const v8bf*)&lA[(16 + lrow) * INc + kb + 16];
        v16bf a0 = cat8(a0l, a0h);
        v16bf a1 = cat8(a1l, a1h);
#pragma unroll
        for (int g = 0; g < 5; ++g) {
            const int ncol = (g * 8 + wave) * 16 + lrow;     // = g*128 + ch
            v16bf bfrag = cat8(*(const v8bf*)(wbase + (size_t)ncol * INc + kb), *(const v8bf*)(wbase + (size_t)ncol * INc + kb + 16));
            acc[0][g] = __builtin_amdgcn_wmma_f32_16x16x32_bf16(
                false, a0, false, bfrag, (short)0, acc[0][g], false, false);
            acc[1][g] = __builtin_amdgcn_wmma_f32_16x16x32_bf16(
                false, a1, false, bfrag, (short)0, acc[1][g], false, false);
        }
    }

    // ---- phase 3: fused gates / state update (all wave-local registers)
    const int ch = wave * 16 + lrow;
    const float bi  = bl[dir * NG +   0 + ch];
    const float bf1 = bl[dir * NG + 128 + ch];
    const float bf2 = bl[dir * NG + 256 + ch];
    const float bg  = bl[dir * NG + 384 + ch];
    const float bo  = bl[dir * NG + 512 + ch];
    const float wci = Wci[dir * HDc + ch];
    const float wf1 = Wf1[dir * HDc + ch];
    const float wf2 = Wf2[dir * HDc + ch];
    const float wco = Wco[dir * HDc + ch];

    float* cdir = cbuf + (size_t)dir * 2 * Wc * Bc * HDc;
    const int ium = (i > 0) ? (i - 1) : 0;
    const float* cup_row  = cdir + (size_t)((ium & 1) * Wc + j) * Bc * HDc;
    const int jlm = (j > 0) ? (j - 1) : 0;
    const float* clf_row  = cdir + (size_t)((i & 1) * Wc + jlm) * Bc * HDc;
    float*       cout_row = cdir + (size_t)((i & 1) * Wc + j) * Bc * HDc;
    __bf16*      hout     = hbuf + ((size_t)dir * HW + i * Wc + j) * Bc * HDc;

#pragma unroll
    for (int mt = 0; mt < 2; ++mt) {
#pragma unroll
        for (int r = 0; r < 8; ++r) {
            const int b = mt * 16 + lg * 8 + r;          // D tile row layout
            const float cu = (i > 0) ? cup_row[b * HDc + ch] : 0.0f;
            const float cl = (j > 0) ? clf_row[b * HDc + ch] : 0.0f;
            const float ig = sigm(acc[mt][0][r] + bi  + wci * (cu + cl));
            const float F1 = sigm(acc[mt][1][r] + bf1 + wf1 * cu);
            const float F2 = sigm(acc[mt][2][r] + bf2 + wf2 * cl);
            const float cn = F1 * cu + F2 * cl + ig * tanhf(acc[mt][3][r] + bg);
            const float o  = sigm(acc[mt][4][r] + bo  + wco * cn);
            const float hn = o * tanhf(cn);
            hout[b * HDc + ch]     = (__bf16)hn;
            cout_row[b * HDc + ch] = cn;
        }
    }
}

// --- final 1x1 conv: (H*W*B x 512) @ (512 x 128) bf16 WMMA GEMM ------------
__global__ __launch_bounds__(256)
void conv_kernel(const __bf16* __restrict__ hbuf, const __bf16* __restrict__ Wcb,
                 const float* __restrict__ bconv, float* __restrict__ out) {
    const int lane = threadIdx.x & 31;
    const int wave = threadIdx.x >> 5;
    const int lrow = lane & 15;
    const int lg   = lane >> 4;
    const int mt   = blockIdx.x * 8 + wave;      // 0..8191 M-tiles
    const int mrow = mt * 16 + lrow;             // A row this lane feeds
    const int ij   = mrow >> 5;
    const int b    = mrow & 31;

    v8f acc[8];
#pragma unroll
    for (int n = 0; n < 8; ++n) acc[n] = (v8f)0.0f;

#pragma unroll
    for (int kt = 0; kt < 16; ++kt) {            // K = 512 = 16 * 32
        const int kb = kt * 32 + lg * 8;
        const int k2 = kb + 16;
        // comb channel k -> (dir = k>>7, ch = k&127); chunks never straddle dirs
        const __bf16* p1 = hbuf + (((size_t)(kb >> 7) * HW + ij) * Bc + b) * HDc + (kb & 127);
        const __bf16* p2 = hbuf + (((size_t)(k2 >> 7) * HW + ij) * Bc + b) * HDc + (k2 & 127);
        v16bf a = cat8(*(const v8bf*)p1, *(const v8bf*)p2);
#pragma unroll
        for (int n = 0; n < 8; ++n) {
            v16bf bfrag = cat8(*(const v8bf*)(Wcb + (size_t)(n * 16 + lrow) * K4 + kb), *(const v8bf*)(Wcb + (size_t)(n * 16 + lrow) * K4 + k2));
            acc[n] = __builtin_amdgcn_wmma_f32_16x16x32_bf16(
                false, a, false, bfrag, (short)0, acc[n], false, false);
        }
    }

#pragma unroll
    for (int n = 0; n < 8; ++n) {
        const int o = n * 16 + lrow;
        const float bcv = bconv[o];
#pragma unroll
        for (int r = 0; r < 8; ++r) {
            const int mg  = mt * 16 + lg * 8 + r;
            const int b2  = mg & 31;
            const int ij2 = mg >> 5;
            const int i2  = ij2 >> 6;
            const int j2  = ij2 & 63;
            out[(((size_t)b2 * OUTc + o) * Hc + i2) * Wc + j2] = acc[n][r] + bcv;
        }
    }
}

// ---------------------------------------------------------------------------
extern "C" void kernel_launch(void* const* d_in, const int* in_sizes, int n_in,
                              void* d_out, int out_size, void* d_ws, size_t ws_size,
                              hipStream_t stream) {
    const float* x      = (const float*)d_in[0];
    const float* W_lin  = (const float*)d_in[1];
    const float* b_lin  = (const float*)d_in[2];
    const float* Wci    = (const float*)d_in[3];
    const float* Wf1    = (const float*)d_in[4];
    const float* Wf2    = (const float*)d_in[5];
    const float* Wco    = (const float*)d_in[6];
    const float* W_conv = (const float*)d_in[7];
    const float* b_conv = (const float*)d_in[8];
    float* out = (float*)d_out;

    // workspace layout (all offsets 256B aligned)
    char* ws = (char*)d_ws;
    __bf16* Wlin_bf  = (__bf16*)(ws);                      // 4*640*288*2 = 1,474,560 B
    __bf16* Wconv_bf = (__bf16*)(ws + 1474560);            // 128*512*2   =   131,072 B
    __bf16* hbuf     = (__bf16*)(ws + 1605632);            // 4*4096*32*128*2 = 134,217,728 B
    float*  cbuf     = (float*) (ws + 135823360);          // 4*2*64*32*128*4 =   8,388,608 B

    // 1) weight conversion to bf16 (stays L2-resident for the whole run)
    {
        const int total = 4 * NG * INc + OUTc * K4;
        prep_kernel<<<(total + 255) / 256, 256, 0, stream>>>(W_lin, W_conv, Wlin_bf, Wconv_bf);
    }

    // 2) 127 wavefront steps over anti-diagonals, all 4 directions per step
    for (int d = 0; d < Hc + Wc - 1; ++d) {
        int imin = d - (Wc - 1); if (imin < 0) imin = 0;
        int imax = (d < Hc - 1) ? d : (Hc - 1);
        int ncells = imax - imin + 1;
        diag_kernel<<<dim3(ncells, 4), 256, 0, stream>>>(
            d, imin, x, Wlin_bf, b_lin, Wci, Wf1, Wf2, Wco, hbuf, cbuf);
    }

    // 3) output projection GEMM + bias
    conv_kernel<<<1024, 256, 0, stream>>>(hbuf, Wconv_bf, b_conv, out);
}
